// GMM2_6176162971960
// MI455X (gfx1250) — matrix operand order repaired and checked by reading, not compiled, not run on hardware
//
#include <hip/hip_runtime.h>
#include <cstdint>
#include <cstddef>

// ---------------- types ----------------
typedef __bf16 bf16_t;
typedef __attribute__((ext_vector_type(16))) __bf16 v16bf;
typedef __attribute__((ext_vector_type(8)))  float  v8f;
typedef __attribute__((ext_vector_type(4)))  unsigned v4u;
typedef __attribute__((ext_vector_type(8)))  int      v8i;
typedef __attribute__((ext_vector_type(4)))  int      v4i;

struct U8pack { unsigned u[8]; };

#define B_N 16384
#define D_N 768
#define K_N 512
#define STAGE_NUM 5
#define LAMBD 0.1f
#define BETA 1.0f
#define EPSF 1e-6f

#define USE_ASYNC_LDS 1
// Tensor Data Mover staging of the B chunk in the fused GEMM.
#if defined(__has_builtin)
#if __has_builtin(__builtin_amdgcn_tensor_load_to_lds)
#define USE_TDM 1
#endif
#endif
#ifndef USE_TDM
#define USE_TDM 0
#endif

// ---- async global->LDS 16B copy (gfx1250 GLOBAL_LOAD_ASYNC_TO_LDS_B128) ----
__device__ inline void async_copy16(void* lds_dst, const void* gsrc) {
#if USE_ASYNC_LDS
  unsigned l = (unsigned)(uintptr_t)lds_dst;
  asm volatile("global_load_async_to_lds_b128 %0, %1, off"
               :: "v"(l), "v"(gsrc)
               : "memory");
#else
  *(uint4*)lds_dst = *(const uint4*)gsrc;
#endif
}
__device__ inline void async_wait() {
#if USE_ASYNC_LDS
  asm volatile("s_wait_asynccnt 0x0" ::: "memory");
#endif
}

#if USE_TDM
// One-descriptor 2D tile DMA: global (row-major, stride_elems) -> LDS with
// 16B padding appended per 64B row (=> LDS row stride 80B = 40 bf16).
// D# layout per CDNA5 ISA ch.8. 16-bit elements; tile_d0 fixed at 32 elems.
__device__ inline void tdm_load_2d_pad(unsigned lds_off, const void* gaddr,
                                       int tile_d1_rows, int stride_elems) {
  unsigned long long ga = (unsigned long long)(uintptr_t)gaddr;
  v4u g0;
  g0[0] = 1u;                                     // count=1 (valid user D#)
  g0[1] = lds_off;                                // lds_addr
  g0[2] = (unsigned)(ga & 0xffffffffu);           // global_addr[31:0]
  g0[3] = (unsigned)((ga >> 32) & 0x01ffffffu)    // global_addr[56:32]
          | (2u << 30);                           // type=2 ("image")
  v8i g1;
  // data_size=1 (2B) | pad_enable | pad_interval=3 (8x8B=64B) | pad_amount=3 (4 DW)
  g1[0] = (int)((1u << 16) | (1u << 20) | (3u << 22) | (3u << 25));
  g1[1] = 0;                                      // abar=0; tensor_dim0[15:0]=0
  g1[2] = 0x10;                                   // tensor_dim0=1<<20; dim1[15:0]=0
  g1[3] = (int)(0x10u | (32u << 16));             // tensor_dim1=1<<20; tile_dim0=32
  g1[4] = tile_d1_rows;                           // tile_dim1; tile_dim2=0
  g1[5] = stride_elems;                           // tensor_dim0_stride[31:0]
  g1[6] = 0;                                      // stride hi; dim1_stride lo
  g1[7] = 0;
  v4i z = {0, 0, 0, 0};
#if __clang_major__ >= 23
  v8i z8 = {0, 0, 0, 0, 0, 0, 0, 0};
  __builtin_amdgcn_tensor_load_to_lds(g0, g1, z, z, z8, 0);
#else
  __builtin_amdgcn_tensor_load_to_lds(g0, g1, z, z, 0);
#endif
}
#endif

// f32 -> bf16, round-to-nearest-even
__device__ inline bf16_t f2bf(float f) {
  unsigned u = __builtin_bit_cast(unsigned, f);
  unsigned r = u + 0x7FFFu + ((u >> 16) & 1u);
  unsigned short h = (unsigned short)(r >> 16);
  return __builtin_bit_cast(bf16_t, h);
}

__device__ inline float wredSum(float v) {
  #pragma unroll
  for (int off = 16; off > 0; off >>= 1) v += __shfl_xor(v, off, 32);
  return v;
}
__device__ inline float wredMax(float v) {
  #pragma unroll
  for (int off = 16; off > 0; off >>= 1) v = fmaxf(v, __shfl_xor(v, off, 32));
  return v;
}

// A-fragment (16x32 bf16): h=lane/16, m=lane%16; v0..3: K=2v+8h, v4..7: K=16+2(v-4)+8h
template <int STRIDE>
__device__ inline v16bf loadAfrag(const bf16_t (*As)[STRIDE], int m, int h) {
  U8pack t;
  #pragma unroll
  for (int v = 0; v < 4; ++v) t.u[v]     = *(const unsigned*)&As[m][2 * v + 8 * h];
  #pragma unroll
  for (int v = 0; v < 4; ++v) t.u[4 + v] = *(const unsigned*)&As[m][16 + 2 * v + 8 * h];
  return __builtin_bit_cast(v16bf, t);
}
// B-fragment (32x16 bf16): n=lane%16, h=lane/16; v0..3: K=2v+8h, v4..7: K=16+2(v-4)+8h
template <int STRIDE>
__device__ inline v16bf loadBfrag(const bf16_t (*Bs)[STRIDE], int n, int h) {
  U8pack t;
  #pragma unroll
  for (int v = 0; v < 8; ++v) t.u[v] = *(const unsigned*)&Bs[n][2 * (v & 3) + 16 * (v >> 2) + 8 * h];
  return __builtin_bit_cast(v16bf, t);
}

// ---------------- fused logits GEMM + row softmax (double-buffered) --------
// block: 16 rows of cls x all 512 prototypes. 8 waves x 4 n-tiles of 16.
#define A_BYTES (16 * 40 * 2)    // 1280
#define B_BYTES (512 * 40 * 2)   // 40960
#define GS_SMEM (2 * A_BYTES + 2 * B_BYTES)  // 84480 (> 16*512*4 logits)
__global__ __launch_bounds__(256) void k_gemm_softmax(
    const bf16_t* __restrict__ clsb,     // [B, D] bf16
    const bf16_t* __restrict__ protob,   // [K, D] bf16 (unit rows)
    float* __restrict__ r_out,           // [B, K] f32 or nullptr
    bf16_t* __restrict__ r_bf)           // [B, K] bf16
{
  __shared__ __align__(16) char smem[GS_SMEM];
  // Compute buffer bases on the fly (no arrays of LDS pointers: clang would
  // emit an addrspacecast static initializer that ld.lld rejects).
  auto Abuf = [&](int b) -> bf16_t(*)[40] {
    return (bf16_t(*)[40])(smem + b * A_BYTES);
  };
  auto Bbuf = [&](int b) -> bf16_t(*)[40] {
    return (bf16_t(*)[40])(smem + 2 * A_BYTES + b * B_BYTES);
  };
  float (*Ls)[K_N] = (float(*)[K_N])smem;  // aliases staging after GEMM

  const int tid  = threadIdx.x;
  const int wave = tid >> 5, lane = tid & 31;
  const int m = lane & 15, h = lane >> 4;
  const int row0 = blockIdx.x * 16;
  // scalar (per-wave-uniform) "am I wave 0" so TDM issue is a real branch
  const bool issuer = (__builtin_amdgcn_readfirstlane((int)tid) < 32);

  v8f acc[4] = {};

  // --- stage helper (issues async copies / TDM into buffer `buf`) ---
  auto stage = [&](int buf, int d0) {
    if (tid < 64) {  // A tile: 16 rows x 32 = 64 x b128
      int rr = tid >> 2, cp = (tid & 3) * 8;
      async_copy16(&Abuf(buf)[rr][cp],
                   clsb + (size_t)(row0 + rr) * D_N + d0 + cp);
    }
#if USE_TDM
    if (issuer)  // one DMA for the whole 512x32 B chunk, padded to stride 40
      tdm_load_2d_pad((unsigned)(uintptr_t)&Bbuf(buf)[0][0], protob + d0,
                      K_N, D_N);
#else
    #pragma unroll
    for (int j = 0; j < 4; ++j) {
      int nrow = (tid >> 1) + j * 128;
      int cp = (tid & 1) * 16;
      const bf16_t* src = protob + (size_t)nrow * D_N + d0 + cp;
      async_copy16(&Bbuf(buf)[nrow][cp], src);
      async_copy16(&Bbuf(buf)[nrow][cp + 8], src + 8);
    }
#endif
  };

  stage(0, 0);  // prologue
  const int NCHUNK = D_N / 32;  // 24
  for (int it = 0; it < NCHUNK; ++it) {
    int cur = it & 1;
#if USE_TDM
    if (issuer) __builtin_amdgcn_s_wait_tensorcnt(0);
#endif
    async_wait();
    __syncthreads();                       // buffer `cur` visible to all
    if (it + 1 < NCHUNK) stage(1 - cur, (it + 1) * 32);  // overlap with math
    v16bf afrag = loadAfrag<40>(Abuf(cur), m, h);
    #pragma unroll
    for (int nt = 0; nt < 4; ++nt) {
      v16bf bfrag = loadBfrag<40>(&Bbuf(cur)[wave * 64 + nt * 16], m, h);
      acc[nt] = __builtin_amdgcn_wmma_f32_16x16x32_bf16(
          false, afrag, false, bfrag, (short)0, acc[nt], false, false);
    }
  }
  __syncthreads();  // staging region dead; safe to alias as logits

  // dump logits: D-matrix lane (row = v + 8h, col = n0 + m)
  #pragma unroll
  for (int nt = 0; nt < 4; ++nt) {
    int n0 = wave * 64 + nt * 16;
    #pragma unroll
    for (int v = 0; v < 8; ++v) Ls[v + 8 * h][n0 + m] = acc[nt][v];
  }
  __syncthreads();

  const float invL = 1.0f / LAMBD;
  for (int rr = wave * 2; rr < wave * 2 + 2; ++rr) {
    float mx = -3.0e38f;
    for (int c = lane; c < K_N; c += 32) mx = fmaxf(mx, Ls[rr][c] * invL);
    mx = wredMax(mx);
    float s = 0.f;
    for (int c = lane; c < K_N; c += 32) {
      float e = __expf(Ls[rr][c] * invL - mx);
      Ls[rr][c] = e;
      s += e;
    }
    s = wredSum(s);
    float inv = 1.0f / s;
    for (int c = lane; c < K_N; c += 32) {
      float p = Ls[rr][c] * inv;
      size_t idx = (size_t)(row0 + rr) * K_N + c;
      if (r_out) r_out[idx] = p;
      r_bf[idx] = f2bf(p);
    }
  }
}

// ---------------- new_proto = r^T @ cls (split over B, f32 atomic accum) ----
__global__ __launch_bounds__(256) void k_proto_update(
    const bf16_t* __restrict__ r_bf,   // [B, K]
    const bf16_t* __restrict__ clsb,   // [B, D]
    float* __restrict__ proto_acc)     // [K, D], pre-zeroed
{
  __shared__ __align__(16) bf16_t As[16][40];   // [k][b]  (A = r^T)
  __shared__ __align__(16) bf16_t Bs[128][40];  // [d][b]
  const int tid = threadIdx.x, wave = tid >> 5, lane = tid & 31;
  const int m = lane & 15, h = lane >> 4;
  const int k0   = blockIdx.x * 16;
  const int dblk = blockIdx.y * 128;
  const int bbeg = blockIdx.z * (B_N / 4);
  const int bend = bbeg + B_N / 4;

  v8f acc = {};
  for (int b0 = bbeg; b0 < bend; b0 += 32) {
    { // A staging: transpose r[b][k] -> As[k][b]
      int bb = tid >> 3, kk = (tid & 7) * 2;
      unsigned pr =
          *(const unsigned*)(r_bf + (size_t)(b0 + bb) * K_N + k0 + kk);
      As[kk][bb]     = __builtin_bit_cast(bf16_t, (unsigned short)(pr & 0xffffu));
      As[kk + 1][bb] = __builtin_bit_cast(bf16_t, (unsigned short)(pr >> 16));
    }
    { // B staging: transpose cls[b][d] -> Bs[d][b]
      int bb = tid >> 3, dd0 = (tid & 7) * 16;
      const bf16_t* src = clsb + (size_t)(b0 + bb) * D_N + dblk + dd0;
      #pragma unroll
      for (int i = 0; i < 16; ++i) Bs[dd0 + i][bb] = src[i];
    }
    __syncthreads();
    v16bf afrag = loadAfrag<40>(As, m, h);
    v16bf bfrag = loadBfrag<40>(&Bs[wave * 16], m, h);
    acc = __builtin_amdgcn_wmma_f32_16x16x32_bf16(
        false, afrag, false, bfrag, (short)0, acc, false, false);
    __syncthreads();
  }
  #pragma unroll
  for (int v = 0; v < 8; ++v) {
    int row = k0 + v + 8 * h;
    int col = dblk + wave * 16 + m;
    atomicAdd(&proto_acc[(size_t)row * D_N + col], acc[v]);
  }
}

// ---------------- row-normalize proto, emit f32 + bf16 ----------------
__global__ __launch_bounds__(256) void k_normalize(
    const float* __restrict__ proto_acc, float* __restrict__ proto_norm,
    bf16_t* __restrict__ protob)
{
  __shared__ float part[8];
  int k = blockIdx.x, tid = threadIdx.x, wave = tid >> 5, lane = tid & 31;
  const float* row = proto_acc + (size_t)k * D_N;
  float s = 0.f;
  for (int d = tid; d < D_N; d += 256) { float p = row[d]; s += p * p; }
  s = wredSum(s);
  if (lane == 0) part[wave] = s;
  __syncthreads();
  if (wave == 0) {
    float t = (lane < 8) ? part[lane] : 0.f;
    t = wredSum(t);
    if (lane == 0) part[0] = 1.0f / (sqrtf(t) + EPSF);
  }
  __syncthreads();
  float inv = part[0];
  for (int d = tid; d < D_N; d += 256) {
    float p = row[d] * inv;
    proto_norm[(size_t)k * D_N + d] = p;
    protob[(size_t)k * D_N + d]     = f2bf(p);
  }
}

// ---------------- z = r @ proto; z_out = beta*z + cls; recon partial -------
__global__ __launch_bounds__(256) void k_recon(
    const bf16_t* __restrict__ r_bf,    // [B, K]
    const bf16_t* __restrict__ protob,  // [K, D]
    const float* __restrict__ cls,      // [B, D] f32
    float* __restrict__ z_out,          // [B, D]
    float* __restrict__ recon_sum)
{
  __shared__ __align__(16) bf16_t As[16][40];   // [b][k]
  __shared__ __align__(16) bf16_t Bs[128][40];  // [d][k]
  __shared__ float part[8];
  const int tid = threadIdx.x, wave = tid >> 5, lane = tid & 31;
  const int m = lane & 15, h = lane >> 4;
  const int row0 = blockIdx.x * 16;
  const int dblk = blockIdx.y * 128;

  v8f acc = {};
  for (int k0 = 0; k0 < K_N; k0 += 32) {
    if (tid < 64) {  // A tile 16x32 via async b128
      int rr = tid >> 2, cp = (tid & 3) * 8;
      async_copy16(&As[rr][cp], r_bf + (size_t)(row0 + rr) * K_N + k0 + cp);
    }
    { // transpose proto[k][d] -> Bs[d][k]
      int kk = tid >> 3, dd0 = (tid & 7) * 16;
      const bf16_t* src = protob + (size_t)(k0 + kk) * D_N + dblk + dd0;
      #pragma unroll
      for (int i = 0; i < 16; ++i) Bs[dd0 + i][kk] = src[i];
    }
    async_wait();
    __syncthreads();
    v16bf afrag = loadAfrag<40>(As, m, h);
    v16bf bfrag = loadBfrag<40>(&Bs[wave * 16], m, h);
    acc = __builtin_amdgcn_wmma_f32_16x16x32_bf16(
        false, afrag, false, bfrag, (short)0, acc, false, false);
    __syncthreads();
  }

  float local = 0.f;
  #pragma unroll
  for (int v = 0; v < 8; ++v) {
    int row = row0 + v + 8 * h;
    int col = dblk + wave * 16 + m;
    size_t idx = (size_t)row * D_N + col;
    float c = cls[idx];
    float z = acc[v];
    z_out[idx] = BETA * z + c;
    float df = z - c;
    local += df * df;
  }
  local = wredSum(local);
  if (lane == 0) part[wave] = local;
  __syncthreads();
  if (wave == 0) {
    float t = (lane < 8) ? part[lane] : 0.f;
    t = wredSum(t);
    if (lane == 0) atomicAdd(recon_sum, t);
  }
}

// ---------------- small kernels ----------------
__global__ void k_cvt(const float* __restrict__ src, bf16_t* __restrict__ dst,
                      size_t n) {
  for (size_t i = (size_t)blockIdx.x * blockDim.x + threadIdx.x; i < n;
       i += (size_t)gridDim.x * blockDim.x)
    dst[i] = f2bf(src[i]);
}

__global__ void k_copy_proto0(const float* __restrict__ src,
                              float* __restrict__ pn, bf16_t* __restrict__ pb,
                              size_t n) {
  for (size_t i = (size_t)blockIdx.x * blockDim.x + threadIdx.x; i < n;
       i += (size_t)gridDim.x * blockDim.x) {
    float v = src[i];
    pn[i] = v;
    pb[i] = f2bf(v);
  }
}

__global__ void k_zero(float* __restrict__ p, size_t n) {
  for (size_t i = (size_t)blockIdx.x * blockDim.x + threadIdx.x; i < n;
       i += (size_t)gridDim.x * blockDim.x)
    p[i] = 0.f;
}

__global__ __launch_bounds__(256) void k_entropy(const float* __restrict__ r,
                                                 float* __restrict__ ent_sum) {
  __shared__ float part[8];
  const int tid = threadIdx.x, wave = tid >> 5, lane = tid & 31;
  size_t n = (size_t)B_N * K_N;
  float s = 0.f;
  for (size_t i = (size_t)blockIdx.x * 256 + tid; i < n;
       i += (size_t)gridDim.x * 256) {
    float p = r[i];
    s += p * __logf(p + EPSF);
  }
  s = wredSum(s);
  if (lane == 0) part[wave] = s;
  __syncthreads();
  if (wave == 0) {
    float t = (lane < 8) ? part[lane] : 0.f;
    t = wredSum(t);
    if (lane == 0) atomicAdd(ent_sum, t);
  }
}

__global__ __launch_bounds__(256) void k_diversity(
    const float* __restrict__ proto, float* __restrict__ div_sum) {
  __shared__ float part[8];
  const int tid = threadIdx.x, wave = tid >> 5, lane = tid & 31;
  int i = blockIdx.y * 16 + (tid >> 4);
  int j = blockIdx.x * 16 + (tid & 15);
  const float* a = proto + (size_t)i * D_N;
  const float* b = proto + (size_t)j * D_N;
  float s = 0.f;
  for (int d = 0; d < D_N; ++d) s += a[d] * b[d];
  float v = (i == j) ? 0.f : fabsf(s);
  v = wredSum(v);
  if (lane == 0) part[wave] = v;
  __syncthreads();
  if (wave == 0) {
    float t = (lane < 8) ? part[lane] : 0.f;
    t = wredSum(t);
    if (lane == 0) atomicAdd(div_sum, t);
  }
}

__global__ void k_finalize(const float* __restrict__ scal,
                           float* __restrict__ out) {
  out[0] = scal[0] / (float)((size_t)B_N * D_N);
  out[1] = -scal[1] / (float)B_N;
  out[2] = scal[2] / (float)((size_t)K_N * K_N);
}

// ---------------- host orchestration ----------------
extern "C" void kernel_launch(void* const* d_in, const int* in_sizes, int n_in,
                              void* d_out, int out_size, void* d_ws,
                              size_t ws_size, hipStream_t stream) {
  const float* cls      = (const float*)d_in[0];   // [B, D]
  const float* proto_in = (const float*)d_in[1];   // [K, D] unit rows

  float* out_r      = (float*)d_out;                   // [B, K]
  float* out_z      = out_r + (size_t)B_N * K_N;       // [B, D]
  float* out_losses = out_z + (size_t)B_N * D_N;       // 3 scalars

  char* p = (char*)d_ws;
  bf16_t* clsb   = (bf16_t*)p; p += (size_t)B_N * D_N * sizeof(bf16_t);
  bf16_t* protob = (bf16_t*)p; p += (size_t)K_N * D_N * sizeof(bf16_t);
  bf16_t* r_bf   = (bf16_t*)p; p += (size_t)B_N * K_N * sizeof(bf16_t);
  float* proto_acc  = (float*)p; p += (size_t)K_N * D_N * sizeof(float);
  float* proto_norm = (float*)p; p += (size_t)K_N * D_N * sizeof(float);
  float* scal       = (float*)p; p += 16 * sizeof(float);

  k_cvt<<<512, 256, 0, stream>>>(cls, clsb, (size_t)B_N * D_N);
  k_copy_proto0<<<128, 256, 0, stream>>>(proto_in, proto_norm, protob,
                                         (size_t)K_N * D_N);

  for (int s = 0; s < STAGE_NUM; ++s) {
    k_gemm_softmax<<<B_N / 16, 256, 0, stream>>>(clsb, protob, nullptr, r_bf);
    k_zero<<<256, 256, 0, stream>>>(proto_acc, (size_t)K_N * D_N);
    k_proto_update<<<dim3(K_N / 16, D_N / 128, 4), 256, 0, stream>>>(
        r_bf, clsb, proto_acc);
    k_normalize<<<K_N, 256, 0, stream>>>(proto_acc, proto_norm, protob);
  }

  k_zero<<<1, 32, 0, stream>>>(scal, 16);
  k_gemm_softmax<<<B_N / 16, 256, 0, stream>>>(clsb, protob, out_r, r_bf);
  k_recon<<<dim3(B_N / 16, D_N / 128), 256, 0, stream>>>(r_bf, protob, cls,
                                                         out_z, &scal[0]);
  k_entropy<<<2048, 256, 0, stream>>>(out_r, &scal[1]);
  k_diversity<<<dim3(K_N / 16, K_N / 16), 256, 0, stream>>>(proto_norm,
                                                            &scal[2]);
  k_finalize<<<1, 1, 0, stream>>>(scal, out_losses);
}
